// SAGEConvCacheReuse_38543036514866
// MI455X (gfx1250) — compile-verified, matrix operand order checked by reading, not run on hardware
//
#include <hip/hip_runtime.h>

// ---------------------------------------------------------------------------
// SAGEConv + cache/reuse for MI455X (gfx1250, wave32, WMMA).
// GEMM runs through v_wmma_f32_16x16x32_f16; aggregation via L2 f32 atomics.
// ---------------------------------------------------------------------------

typedef __attribute__((ext_vector_type(8)))  _Float16 v8h;
typedef __attribute__((ext_vector_type(16))) _Float16 v16h;
typedef __attribute__((ext_vector_type(8)))  float    v8f;

#define D 128          // feature dim (both in and out), fixed by the problem
#define TILE 16

__device__ __forceinline__ v16h cat8(v8h lo, v8h hi) {
    return __builtin_shufflevector(lo, hi, 0,1,2,3,4,5,6,7,8,9,10,11,12,13,14,15);
}

// Load 16 halves for one WMMA operand register set: 8 contiguous halves
// (VGPR0-3: K = base..base+7) plus 8 more at +16 halves (VGPR4-7: K+16..K+23).
__device__ __forceinline__ v16h load_op16(const _Float16* p) {
    v8h lo = *(const v8h*)(p);
    v8h hi = *(const v8h*)(p + 16);
    return cat8(lo, hi);
}

// ---------------------------------------------------------------------------
// 1) zero scratch accumulators (ssum: n_dst*D floats, then deg: n_dst floats)
// ---------------------------------------------------------------------------
__global__ void k_zero(float* p, long long n) {
    long long i = (long long)blockIdx.x * blockDim.x + threadIdx.x;
    if (i < n) p[i] = 0.0f;
}

// ---------------------------------------------------------------------------
// 2) edge aggregation: wave per edge, lane owns 4 features (float4 gather,
//    4 global_atomic_add_f32 scatter). lane0 counts the degree.
// ---------------------------------------------------------------------------
__global__ void k_aggregate(const float* __restrict__ feat_src,
                            const int* __restrict__ src,
                            const int* __restrict__ dst,
                            float* __restrict__ ssum,
                            float* __restrict__ deg,
                            int n_edges) {
    int e    = blockIdx.x * (blockDim.x >> 5) + (threadIdx.x >> 5);
    int lane = threadIdx.x & 31;
    if (e >= n_edges) return;
    int s = src[e];
    int d = dst[e];
    float4 v = *(const float4*)(feat_src + (size_t)s * D + lane * 4);
    float* out = ssum + (size_t)d * D + lane * 4;
    atomicAdd(out + 0, v.x);
    atomicAdd(out + 1, v.y);
    atomicAdd(out + 2, v.z);
    atomicAdd(out + 3, v.w);
    if (lane == 0) atomicAdd(deg + d, 1.0f);
}

// ---------------------------------------------------------------------------
// 3) mean-normalize + convert both GEMM activation matrices to f16
//    featA = feat_dst (= feat_src[:n_dst]) as f16
//    featB = h_neigh  (= ssum/deg, 0 if deg==0) as f16
// ---------------------------------------------------------------------------
__global__ void k_finalize(const float* __restrict__ feat_src,
                           const float* __restrict__ ssum,
                           const float* __restrict__ deg,
                           _Float16* __restrict__ featA,
                           _Float16* __restrict__ featB,
                           int n_dst) {
    int row  = blockIdx.x * (blockDim.x >> 5) + (threadIdx.x >> 5);
    int lane = threadIdx.x & 31;
    if (row >= n_dst) return;
    float dg  = deg[row];
    float inv = dg > 0.0f ? 1.0f / fmaxf(dg, 1.0f) : 0.0f;
    size_t base = (size_t)row * D + lane * 4;
    float4 fv = *(const float4*)(feat_src + base);
    float4 sv = *(const float4*)(ssum + base);
    featA[base + 0] = (_Float16)fv.x;
    featA[base + 1] = (_Float16)fv.y;
    featA[base + 2] = (_Float16)fv.z;
    featA[base + 3] = (_Float16)fv.w;
    featB[base + 0] = (_Float16)(sv.x * inv);
    featB[base + 1] = (_Float16)(sv.y * inv);
    featB[base + 2] = (_Float16)(sv.z * inv);
    featB[base + 3] = (_Float16)(sv.w * inv);
}

// ---------------------------------------------------------------------------
// 4) convert both 128x128 weight matrices to f16 (row-major, W[n][k])
// ---------------------------------------------------------------------------
__global__ void k_cvt_weights(const float* __restrict__ Ws,
                              const float* __restrict__ Wn,
                              _Float16* __restrict__ hWs,
                              _Float16* __restrict__ hWn,
                              int n) {
    int i = blockIdx.x * blockDim.x + threadIdx.x;
    if (i < n) {
        hWs[i] = (_Float16)Ws[i];
        hWn[i] = (_Float16)Wn[i];
    }
}

// ---------------------------------------------------------------------------
// 5) fused dual-GEMM epilogue-free WMMA kernel:
//    rst[m][n] = sum_k featA[m][k]*Ws[n][k] + sum_k featB[m][k]*Wn[n][k]
//    Wave computes one 16x16 output tile; block (8 waves) covers all 128 cols
//    for one 16-row stripe. A operand: row per lane (lanes16-31 take K+8 /
//    K+24 halves per the ISA layout). B columns of W^T are W rows, so both
//    operands load contiguously from row-major storage.
// ---------------------------------------------------------------------------
__global__ void k_wmma_gemm(const _Float16* __restrict__ featA,
                            const _Float16* __restrict__ featB,
                            const _Float16* __restrict__ Ws,
                            const _Float16* __restrict__ Wn,
                            float* __restrict__ rst,
                            int n_dst) {
    const int lane = threadIdx.x & 31;
    const int wave = threadIdx.x >> 5;     // 0..7 -> column tile
    const int mt   = blockIdx.x * TILE;
    const int nt   = wave * TILE;
    const int l15  = lane & 15;
    const int hp   = lane >> 4;            // lane half selects K phase

    int rowA = mt + l15;
    if (rowA >= n_dst) rowA = n_dst - 1;   // clamp: EXEC must stay all-1s

    const _Float16* a0 = featA + (size_t)rowA * D + hp * 8;
    const _Float16* a1 = featB + (size_t)rowA * D + hp * 8;
    const _Float16* b0 = Ws + (size_t)(nt + l15) * D + hp * 8;
    const _Float16* b1 = Wn + (size_t)(nt + l15) * D + hp * 8;

    v8f c = {};
#pragma unroll
    for (int k0 = 0; k0 < D; k0 += 32) {
        v16h As = load_op16(a0 + k0);
        v16h Bs = load_op16(b0 + k0);
        c = __builtin_amdgcn_wmma_f32_16x16x32_f16(false, As, false, Bs,
                                                   (short)0, c, false, false);
        v16h An = load_op16(a1 + k0);
        v16h Bn = load_op16(b1 + k0);
        c = __builtin_amdgcn_wmma_f32_16x16x32_f16(false, An, false, Bn,
                                                   (short)0, c, false, false);
    }

    // C/D layout: VGPR r -> M = r + 8*(lane>=16), N = lane&15
#pragma unroll
    for (int r = 0; r < 8; ++r) {
        int row = mt + r + hp * 8;
        if (row < n_dst) rst[(size_t)row * D + nt + l15] = c[r];
    }
}

// ---------------------------------------------------------------------------
// 6) assemble `full`: wave per row p; binary search sorted reuse_indices for
//    lower_bound(p). Hit -> reuse_embedding[rank]; miss -> rst[p - rank].
// ---------------------------------------------------------------------------
__global__ void k_assemble(const float* __restrict__ rst,
                           const float* __restrict__ reuse_emb,
                           const int* __restrict__ reuse_idx,
                           int n_reuse,
                           float* __restrict__ full,
                           int full_len) {
    int p    = blockIdx.x * (blockDim.x >> 5) + (threadIdx.x >> 5);
    int lane = threadIdx.x & 31;
    if (p >= full_len) return;
    int lo = 0, hi = n_reuse;
    while (lo < hi) {                    // uniform across the wave -> scalar
        int mid = (lo + hi) >> 1;
        if (reuse_idx[mid] < p) lo = mid + 1; else hi = mid;
    }
    const float* srcp;
    if (lo < n_reuse && reuse_idx[lo] == p) srcp = reuse_emb + (size_t)lo * D;
    else                                    srcp = rst + (size_t)(p - lo) * D;
    float4 v = *(const float4*)(srcp + lane * 4);
    *(float4*)(full + (size_t)p * D + lane * 4) = v;
}

// ---------------------------------------------------------------------------
// 7) cache gather: wave per cached row
// ---------------------------------------------------------------------------
__global__ void k_cache(const float* __restrict__ full,
                        const int* __restrict__ cache_idx,
                        float* __restrict__ out_cache,
                        int n_cache) {
    int i    = blockIdx.x * (blockDim.x >> 5) + (threadIdx.x >> 5);
    int lane = threadIdx.x & 31;
    if (i >= n_cache) return;
    int p = cache_idx[i];
    float4 v = *(const float4*)(full + (size_t)p * D + lane * 4);
    *(float4*)(out_cache + (size_t)i * D + lane * 4) = v;
}

// ---------------------------------------------------------------------------
extern "C" void kernel_launch(void* const* d_in, const int* in_sizes, int n_in,
                              void* d_out, int out_size, void* d_ws, size_t ws_size,
                              hipStream_t stream) {
    const float* feat_src  = (const float*)d_in[0];
    const float* W_self    = (const float*)d_in[1];
    const float* W_neigh   = (const float*)d_in[2];
    const float* reuse_emb = (const float*)d_in[3];
    const int*   src       = (const int*)d_in[4];
    const int*   dst       = (const int*)d_in[5];
    const int*   reuse_idx = (const int*)d_in[6];
    const int*   cache_idx = (const int*)d_in[7];
    // d_in[8] is n_dst on device; recover it on host from sizes instead:
    const int n_edges  = in_sizes[4];
    const int n_reuse  = in_sizes[6];
    const int n_cache  = in_sizes[7];
    const int n_dst    = out_size / D - n_reuse - n_cache;
    const int full_len = n_dst + n_reuse;

    // ---- workspace layout (256B aligned chunks) ----
    auto align256 = [](size_t x) { return (x + 255) & ~(size_t)255; };
    char*  ws   = (char*)d_ws;
    size_t off  = 0;
    float* ssum = (float*)(ws + off);          // n_dst*D f32; reused as `rst`
    off += align256((size_t)n_dst * D * sizeof(float));
    float* deg  = (float*)(ws + off);
    off += align256((size_t)n_dst * sizeof(float));
    _Float16* featA = (_Float16*)(ws + off);   // feat_dst as f16
    off += align256((size_t)n_dst * D * sizeof(_Float16));
    _Float16* featB = (_Float16*)(ws + off);   // h_neigh as f16
    off += align256((size_t)n_dst * D * sizeof(_Float16));
    _Float16* hWs = (_Float16*)(ws + off);
    off += align256((size_t)D * D * sizeof(_Float16));
    _Float16* hWn = (_Float16*)(ws + off);
    off += align256((size_t)D * D * sizeof(_Float16));
    (void)ws_size; (void)n_in;

    float* full      = (float*)d_out;                       // [full_len, D]
    float* out_cache = (float*)d_out + (size_t)full_len * D; // [n_cache, D]
    float* rst       = ssum;  // safe: ssum fully consumed before GEMM writes

    const int BLK = 256;              // 8 wave32s per block
    const int WPB = BLK / 32;         // waves (rows/edges) per block

    // 1) zero accumulators (ssum, then deg, one k_zero launch each)
    {
        long long nz = (long long)n_dst * D;
        k_zero<<<(int)((nz + BLK - 1) / BLK), BLK, 0, stream>>>(ssum, nz);
        k_zero<<<(n_dst + BLK - 1) / BLK, BLK, 0, stream>>>(deg, n_dst);
    }
    // 2) edge aggregation
    k_aggregate<<<(n_edges + WPB - 1) / WPB, BLK, 0, stream>>>(
        feat_src, src, dst, ssum, deg, n_edges);
    // 3) mean + f16 conversion of activations
    k_finalize<<<(n_dst + WPB - 1) / WPB, BLK, 0, stream>>>(
        feat_src, ssum, deg, featA, featB, n_dst);
    // 4) f16 conversion of weights
    k_cvt_weights<<<(D * D + BLK - 1) / BLK, BLK, 0, stream>>>(
        W_self, W_neigh, hWs, hWn, D * D);
    // 5) fused dual-GEMM via WMMA (one block per 16-row stripe)
    k_wmma_gemm<<<(n_dst + TILE - 1) / TILE, BLK, 0, stream>>>(
        featA, featB, hWs, hWn, rst, n_dst);
    // 6) assemble full output with reuse rows spliced in
    k_assemble<<<(full_len + WPB - 1) / WPB, BLK, 0, stream>>>(
        rst, reuse_emb, reuse_idx, n_reuse, full, full_len);
    // 7) cache gather
    k_cache<<<(n_cache + WPB - 1) / WPB, BLK, 0, stream>>>(
        full, cache_idx, out_cache, n_cache);
}
